// FeedbackTransformerKV_79061757985042
// MI455X (gfx1250) — matrix operand order repaired and checked by reading, not compiled, not run on hardware
//
#include <hip/hip_runtime.h>
#include <hip/hip_bf16.h>

typedef __attribute__((ext_vector_type(8)))  __bf16 v8bf;
typedef __attribute__((ext_vector_type(16))) __bf16 v16bf;
typedef __attribute__((ext_vector_type(8)))  float  v8f;

#define S_T 128
#define B_T 16
#define D_T 512
#define H_T 8
#define DK_T 64
#define DFF_T 2048
#define L_T 4
#define P_T 4096

#define NWG 8
#define TPB 512
#define GSZ (NWG * TPB)
#define WPW 16  // waves per WG

// ---- workspace byte offsets (all 256-aligned) ----
#define OFF_BAR    ((size_t)0)
#define OFF_COMB   ((size_t)256)
#define OFF_WQT    ((size_t)1024)
#define OFF_WOT    (OFF_WQT   + (size_t)L_T*D_T*D_T*2)
#define OFF_WFF1T  (OFF_WOT   + (size_t)L_T*D_T*D_T*2)
#define OFF_WFF2T  (OFF_WFF1T + (size_t)L_T*D_T*DFF_T*2)
#define OFF_WKT    (OFF_WFF2T + (size_t)L_T*DFF_T*D_T*2)
#define OFF_WVT    (OFF_WKT   + (size_t)D_T*D_T*2)
#define OFF_MEMK   (OFF_WVT   + (size_t)D_T*D_T*2)
#define OFF_MEMV   (OFF_MEMK  + (size_t)S_T*B_T*D_T*4)
#define OFF_XBUF   (OFF_MEMV  + (size_t)S_T*B_T*D_T*4)
#define OFF_QBUF   (OFF_XBUF  + (size_t)B_T*D_T*4)
#define OFF_AVBF   (OFF_QBUF  + (size_t)B_T*D_T*4)
#define OFF_HBF    (OFF_AVBF  + (size_t)B_T*D_T*2)
#define OFF_OUTS   (OFF_HBF   + (size_t)B_T*DFF_T*2)

__device__ __forceinline__ unsigned short f2bf(float f) {
  unsigned u = __float_as_uint(f);
  u += 0x7FFFu + ((u >> 16) & 1u);          // round-to-nearest-even
  return (unsigned short)(u >> 16);
}

__device__ __forceinline__ float wred_sum(float v) {
  #pragma unroll
  for (int o = 16; o; o >>= 1) v += __shfl_xor(v, o, 32);
  return v;
}
__device__ __forceinline__ float wred_max(float v) {
  #pragma unroll
  for (int o = 16; o; o >>= 1) v = fmaxf(v, __shfl_xor(v, o, 32));
  return v;
}

// device-wide generation barrier (all NWG workgroups resident)
__device__ __forceinline__ void gsync(unsigned* bar) {
  __threadfence();
  __syncthreads();
  if (threadIdx.x == 0) {
    unsigned old = __hip_atomic_fetch_add(bar, 1u, __ATOMIC_ACQ_REL, __HIP_MEMORY_SCOPE_AGENT);
    unsigned target = (old / NWG + 1u) * NWG;
    while (__hip_atomic_load(bar, __ATOMIC_ACQUIRE, __HIP_MEMORY_SCOPE_AGENT) < target)
      __builtin_amdgcn_s_sleep(2);
  }
  __syncthreads();
  __threadfence();
}

#define SHUF16(lo, hi) __builtin_shufflevector(lo, hi, 0,1,2,3,4,5,6,7,8,9,10,11,12,13,14,15)

// One 16x16 output tile of [16 x K] @ [K x N]; B pre-transposed ([N][K] bf16).
// K is processed in groups of 128 (4 WMMA chunks). B fragments for group g+1
// are loaded into a second register set BEFORE the 4 WMMAs of group g, so the
// scheduler can use partial s_wait_loadcnt and overlap L2 latency with the
// matrix pipe instead of draining loadcnt to zero per chunk.
template <int K>
__device__ __forceinline__ v8f wmma_rowtile(const unsigned short* A, int lda,
                                            const unsigned short* BT, int Ncol, int lane) {
  constexpr int NG = K / 128;
  v8f acc = {};
  const int half = lane >> 4;
  const unsigned short* arow = A + (size_t)(lane & 15) * lda;
  const unsigned short* brow = BT + (size_t)Ncol * K + half * 16;
  v16bf b0 = *(const v16bf*)(brow + 0);
  v16bf b1 = *(const v16bf*)(brow + 32);
  v16bf b2 = *(const v16bf*)(brow + 64);
  v16bf b3 = *(const v16bf*)(brow + 96);
  #pragma unroll
  for (int g = 0; g < NG; ++g) {
    const int kg = g * 128;
    v16bf n0, n1, n2, n3;
    if (g + 1 < NG) {                       // prefetch next group's B fragments
      const unsigned short* nb = brow + kg + 128;
      n0 = *(const v16bf*)(nb + 0);
      n1 = *(const v16bf*)(nb + 32);
      n2 = *(const v16bf*)(nb + 64);
      n3 = *(const v16bf*)(nb + 96);
    }
    {
      v8bf alo = *(const v8bf*)(arow + kg + half * 16);
      v8bf ahi = *(const v8bf*)(arow + kg + 8 + half * 16);
      acc = __builtin_amdgcn_wmma_f32_16x16x32_bf16(false, SHUF16(alo, ahi), false, b0,
                                                    (short)0, acc, false, false);
    }
    {
      v8bf alo = *(const v8bf*)(arow + kg + 32 + half * 16);
      v8bf ahi = *(const v8bf*)(arow + kg + 40 + half * 16);
      acc = __builtin_amdgcn_wmma_f32_16x16x32_bf16(false, SHUF16(alo, ahi), false, b1,
                                                    (short)0, acc, false, false);
    }
    {
      v8bf alo = *(const v8bf*)(arow + kg + 64 + half * 16);
      v8bf ahi = *(const v8bf*)(arow + kg + 72 + half * 16);
      acc = __builtin_amdgcn_wmma_f32_16x16x32_bf16(false, SHUF16(alo, ahi), false, b2,
                                                    (short)0, acc, false, false);
    }
    {
      v8bf alo = *(const v8bf*)(arow + kg + 96 + half * 16);
      v8bf ahi = *(const v8bf*)(arow + kg + 104 + half * 16);
      acc = __builtin_amdgcn_wmma_f32_16x16x32_bf16(false, SHUF16(alo, ahi), false, b3,
                                                    (short)0, acc, false, false);
    }
    if (g + 1 < NG) { b0 = n0; b1 = n1; b2 = n2; b3 = n3; }
  }
  return acc;
}

// convert fp32 [K][N] -> bf16 transposed [N][K]
__device__ __forceinline__ void cvt_tr(const float* src, unsigned short* dst,
                                       int K, int N, int gtid) {
  int total = K * N;
  for (int i = gtid; i < total; i += GSZ) {
    int n = i / K, k = i - n * K;
    dst[i] = f2bf(src[(size_t)k * N + n]);
  }
}

__global__ __launch_bounds__(TPB, 1) void feedback_transformer_kernel(
    const float* __restrict__ x_seq, const float* __restrict__ w_query,
    const float* __restrict__ query_pos_bias, const float* __restrict__ key_pos_emb,
    const float* __restrict__ w_out, const float* __restrict__ b_out,
    const float* __restrict__ ln1_g, const float* __restrict__ ln1_b,
    const float* __restrict__ ln2_g, const float* __restrict__ ln2_b,
    const float* __restrict__ w_ff1, const float* __restrict__ b_ff1,
    const float* __restrict__ w_ff2, const float* __restrict__ b_ff2,
    const float* __restrict__ norm_g, const float* __restrict__ norm_b,
    const float* __restrict__ comb_w, const float* __restrict__ w_key,
    const float* __restrict__ w_value, float* __restrict__ out, char* __restrict__ ws) {
  const int tid = threadIdx.x;
  const int wg = blockIdx.x;
  const int gtid = wg * TPB + tid;
  const int lane = tid & 31;
  const int wavein = tid >> 5;               // 0..15
  const int gjob = wavein * NWG + wg;        // balanced wave job id, 0..127

  unsigned*       bar   = (unsigned*)(ws + OFF_BAR);
  float*          comb  = (float*)(ws + OFF_COMB);
  unsigned short* wqT   = (unsigned short*)(ws + OFF_WQT);
  unsigned short* woT   = (unsigned short*)(ws + OFF_WOT);
  unsigned short* wf1T  = (unsigned short*)(ws + OFF_WFF1T);
  unsigned short* wf2T  = (unsigned short*)(ws + OFF_WFF2T);
  unsigned short* wkT   = (unsigned short*)(ws + OFF_WKT);
  unsigned short* wvT   = (unsigned short*)(ws + OFF_WVT);
  float*          memk  = (float*)(ws + OFF_MEMK);
  float*          memv  = (float*)(ws + OFF_MEMV);
  float*          xbuf  = (float*)(ws + OFF_XBUF);
  float*          qbuf  = (float*)(ws + OFF_QBUF);
  unsigned short* avbf  = (unsigned short*)(ws + OFF_AVBF);
  unsigned short* hbf   = (unsigned short*)(ws + OFF_HBF);
  float*          outs  = (float*)(ws + OFF_OUTS);

  __shared__ unsigned short smemA[B_T * D_T];   // 16 KiB bf16 activation tile

  // redundant per-WG LayerNorm of xbuf straight into LDS (one wave per row;
  // every WG computes all 16 rows, so no device-wide sync is needed after)
  auto ln_to_lds = [&](const float* g, const float* b) {
    const float* xr = xbuf + wavein * D_T;
    float s = 0.f, sq = 0.f;
    for (int c = lane; c < D_T; c += 32) { float v = xr[c]; s += v; sq += v * v; }
    s = wred_sum(s); sq = wred_sum(sq);
    float mu = s * (1.0f / D_T);
    float rstd = rsqrtf(sq * (1.0f / D_T) - mu * mu + 1e-5f);
    for (int c = lane; c < D_T; c += 32)
      smemA[wavein * D_T + c] = f2bf((xr[c] - mu) * rstd * g[c] + b[c]);
  };

  // ---------------- Phase A: weight bf16 conversion + transpose (once) ----------
  for (int l = 0; l < L_T; ++l) {
    cvt_tr(w_query + (size_t)l * D_T * D_T,  wqT  + (size_t)l * D_T * D_T,  D_T,  D_T,  gtid);
    cvt_tr(w_out   + (size_t)l * D_T * D_T,  woT  + (size_t)l * D_T * D_T,  D_T,  D_T,  gtid);
    cvt_tr(w_ff1   + (size_t)l * D_T * DFF_T, wf1T + (size_t)l * D_T * DFF_T, D_T, DFF_T, gtid);
    cvt_tr(w_ff2   + (size_t)l * DFF_T * D_T, wf2T + (size_t)l * DFF_T * D_T, DFF_T, D_T, gtid);
  }
  cvt_tr(w_key,   wkT, D_T, D_T, gtid);
  cvt_tr(w_value, wvT, D_T, D_T, gtid);
  if (gtid == 0) {   // softmax over L+1 combine weights
    float m = -1e30f;
    for (int i = 0; i < L_T + 1; ++i) m = fmaxf(m, comb_w[i]);
    float s = 0.f, e[L_T + 1];
    for (int i = 0; i < L_T + 1; ++i) { e[i] = __expf(comb_w[i] - m); s += e[i]; }
    for (int i = 0; i < L_T + 1; ++i) comb[i] = e[i] / s;
  }
  gsync(bar);

  // ---------------- sequential scan over time steps ------------------------------
  for (int t = 0; t < S_T; ++t) {
    // load x = x_seq[t]; outs[0] = x
    for (int i = gtid; i < B_T * D_T; i += GSZ) {
      float v = x_seq[(size_t)t * B_T * D_T + i];
      xbuf[i] = v;
      outs[i] = v;
    }
    gsync(bar);

    for (int l = 0; l < L_T; ++l) {
      if (t > 0) {
        // ---- LN1 (per-WG, into LDS) then q = z @ Wq : 32 N-tiles ----
        ln_to_lds(ln1_g + l * D_T, ln1_b + l * D_T);
        __syncthreads();
        for (int ti = gjob; ti < D_T / 16; ti += NWG * WPW) {
          int N = ti * 16 + (lane & 15);
          v8f acc = wmma_rowtile<D_T>(smemA, D_T, wqT + (size_t)l * D_T * D_T, N, lane);
          int mb = (lane >> 4) * 8;
          #pragma unroll
          for (int r = 0; r < 8; ++r) qbuf[(mb + r) * D_T + N] = acc[r];
        }
        gsync(bar);

        // ---- relative-position attention: one wave per (b,h) pair ----
        {
          int b = gjob >> 3, h = gjob & 7;
          int d0 = lane * 2;
          float2 qv = *(const float2*)(qbuf + b * D_T + h * DK_T + d0);
          float2 pb = *(const float2*)(query_pos_bias + (size_t)l * H_T * DK_T + h * DK_T + d0);
          float qk0 = qv.x + pb.x, qk1 = qv.y + pb.y;
          float sc0 = -1e9f, sc1 = -1e9f, sc2 = -1e9f, sc3 = -1e9f;
          for (int j = 0; j < t; ++j) {
            float2 mk = *(const float2*)(memk + ((size_t)j * B_T + b) * D_T + h * DK_T + d0);
            int row = P_T - t + j;  // in [P-t, P-1], always valid here
            float2 kp = *(const float2*)(key_pos_emb +
                          (((size_t)l * P_T + row) * H_T + h) * DK_T + d0);
            float part = qk0 * mk.x + qk1 * mk.y + qv.x * kp.x + qv.y * kp.y;
            float sc = wred_sum(part) * 0.125f;   // 1/sqrt(64)
            int slot = j >> 5;
            if ((j & 31) == lane) {
              if (slot == 0) sc0 = sc; else if (slot == 1) sc1 = sc;
              else if (slot == 2) sc2 = sc; else sc3 = sc;
            }
          }
          float m = wred_max(fmaxf(fmaxf(sc0, sc1), fmaxf(sc2, sc3)));
          float e0 = __expf(sc0 - m), e1 = __expf(sc1 - m);
          float e2 = __expf(sc2 - m), e3 = __expf(sc3 - m);
          float rs = 1.0f / wred_sum(e0 + e1 + e2 + e3);
          float av0 = 0.f, av1 = 0.f;
          for (int j = 0; j < t; ++j) {
            int slot = j >> 5;
            float es = slot == 0 ? e0 : slot == 1 ? e1 : slot == 2 ? e2 : e3;
            float wj = __shfl(es, j & 31, 32) * rs;
            float2 mv = *(const float2*)(memv + ((size_t)j * B_T + b) * D_T + h * DK_T + d0);
            av0 += wj * mv.x; av1 += wj * mv.y;
          }
          unsigned pk = (unsigned)f2bf(av0) | ((unsigned)f2bf(av1) << 16);
          *(unsigned*)(avbf + b * D_T + h * DK_T + d0) = pk;
        }
        gsync(bar);

        // ---- x += av @ Wout + b_out (stage avbf into LDS) ----
        for (int i = tid; i < B_T * D_T / 2; i += TPB)
          ((unsigned*)smemA)[i] = ((const unsigned*)avbf)[i];
        __syncthreads();
        for (int ti = gjob; ti < D_T / 16; ti += NWG * WPW) {
          int N = ti * 16 + (lane & 15);
          v8f acc = wmma_rowtile<D_T>(smemA, D_T, woT + (size_t)l * D_T * D_T, N, lane);
          int mb = (lane >> 4) * 8;
          float bo = b_out[l * D_T + N];
          #pragma unroll
          for (int r = 0; r < 8; ++r) xbuf[(mb + r) * D_T + N] += acc[r] + bo;
        }
        gsync(bar);
      }

      // ---- LN2 (per-WG, into LDS) then h = relu(z @ Wff1 + b1), 128 tiles ----
      ln_to_lds(ln2_g + l * D_T, ln2_b + l * D_T);
      __syncthreads();
      for (int ti = gjob; ti < DFF_T / 16; ti += NWG * WPW) {
        int N = ti * 16 + (lane & 15);
        v8f acc = wmma_rowtile<D_T>(smemA, D_T, wf1T + (size_t)l * D_T * DFF_T, N, lane);
        int mb = (lane >> 4) * 8;
        float bb = b_ff1[l * DFF_T + N];
        #pragma unroll
        for (int r = 0; r < 8; ++r)
          hbf[(mb + r) * DFF_T + N] = f2bf(fmaxf(acc[r] + bb, 0.0f));
      }
      gsync(bar);

      // ---- x += h @ Wff2 + b2 ; outs[l+1] = x ----
      for (int ti = gjob; ti < D_T / 16; ti += NWG * WPW) {
        int N = ti * 16 + (lane & 15);
        v8f acc = wmma_rowtile<DFF_T>(hbf, DFF_T, wf2T + (size_t)l * DFF_T * D_T, N, lane);
        int mb = (lane >> 4) * 8;
        float bb = b_ff2[l * D_T + N];
        #pragma unroll
        for (int r = 0; r < 8; ++r) {
          int idx = (mb + r) * D_T + N;
          float nx = xbuf[idx] + acc[r] + bb;
          xbuf[idx] = nx;
          outs[(size_t)(l + 1) * B_T * D_T + idx] = nx;
        }
      }
      gsync(bar);
    }  // layers

    // ---- fused epilogue: combine -> LDS (per-WG), final LN -> out[t],
    //      then mem_k[t]/mem_v[t] = mem @ Wk/Wv (64 tiles) ----
    for (int i = tid; i < B_T * D_T; i += TPB) {
      float acc = 0.f;
      #pragma unroll
      for (int k = 0; k < L_T + 1; ++k) acc += comb[k] * outs[(size_t)k * B_T * D_T + i];
      smemA[i] = f2bf(acc);
    }
    if (gjob >= 64 && gjob < 64 + B_T) {     // final LayerNorm rows (disjoint waves)
      int row = gjob - 64;
      const float* xr = xbuf + row * D_T;
      float s = 0.f, sq = 0.f;
      for (int c = lane; c < D_T; c += 32) { float v = xr[c]; s += v; sq += v * v; }
      s = wred_sum(s); sq = wred_sum(sq);
      float mu = s * (1.0f / D_T);
      float rstd = rsqrtf(sq * (1.0f / D_T) - mu * mu + 1e-5f);
      for (int c = lane; c < D_T; c += 32)
        out[(size_t)t * B_T * D_T + row * D_T + c] =
            (xr[c] - mu) * rstd * norm_g[c] + norm_b[c];
    }
    __syncthreads();
    for (int job = gjob; job < 64; job += NWG * WPW) {
      int isv = job >= 32;
      int ti = job & 31;
      int N = ti * 16 + (lane & 15);
      const unsigned short* WT = isv ? wvT : wkT;
      float* dst = (isv ? memv : memk) + (size_t)t * B_T * D_T;
      v8f acc = wmma_rowtile<D_T>(smemA, D_T, WT, N, lane);
      int mb = (lane >> 4) * 8;
      #pragma unroll
      for (int r = 0; r < 8; ++r) dst[(mb + r) * D_T + N] = acc[r];
    }
    gsync(bar);
  }  // time steps
}

extern "C" void kernel_launch(void* const* d_in, const int* in_sizes, int n_in,
                              void* d_out, int out_size, void* d_ws, size_t ws_size,
                              hipStream_t stream) {
  (void)in_sizes; (void)n_in; (void)out_size; (void)ws_size;
  const float* x_seq          = (const float*)d_in[0];
  const float* w_query        = (const float*)d_in[1];
  const float* query_pos_bias = (const float*)d_in[2];
  const float* key_pos_emb    = (const float*)d_in[3];
  const float* w_out          = (const float*)d_in[4];
  const float* b_out          = (const float*)d_in[5];
  const float* ln1_g          = (const float*)d_in[6];
  const float* ln1_b          = (const float*)d_in[7];
  const float* ln2_g          = (const float*)d_in[8];
  const float* ln2_b          = (const float*)d_in[9];
  const float* w_ff1          = (const float*)d_in[10];
  const float* b_ff1          = (const float*)d_in[11];
  const float* w_ff2          = (const float*)d_in[12];
  const float* b_ff2          = (const float*)d_in[13];
  const float* norm_g         = (const float*)d_in[14];
  const float* norm_b         = (const float*)d_in[15];
  const float* comb_w         = (const float*)d_in[16];
  const float* w_key          = (const float*)d_in[17];
  const float* w_value        = (const float*)d_in[18];

  hipMemsetAsync(d_ws, 0, 256, stream);  // reset global barrier counter
  hipLaunchKernelGGL(feedback_transformer_kernel, dim3(NWG), dim3(TPB), 0, stream,
                     x_seq, w_query, query_pos_bias, key_pos_emb, w_out, b_out,
                     ln1_g, ln1_b, ln2_g, ln2_b, w_ff1, b_ff1, w_ff2, b_ff2,
                     norm_g, norm_b, comb_w, w_key, w_value,
                     (float*)d_out, (char*)d_ws);
}
